// Generator_81312320848270
// MI455X (gfx1250) — matrix operand order repaired and checked by reading, not compiled, not run on hardware
//
#include <hip/hip_runtime.h>
#include <hip/hip_bf16.h>

// ---------------- problem constants (from reference) ----------------
static const int  N_   = 16000;
static const int  F_   = 128;
static const int  E_   = 256000;
static const int  G_   = 512;
static const long NNZ_ = 512000;

typedef __attribute__((ext_vector_type(16))) _Float16 v16h;
typedef __attribute__((ext_vector_type(8)))  _Float16 v8h;
typedef __attribute__((ext_vector_type(8)))  float    v8f;

// ---------------- device helpers ----------------
__device__ __forceinline__ float sigf(float x) {
    return 1.0f / (1.0f + __expf(-x));
}

// A fragment: 16x32 f16 tile from row-major A (stride lda), contiguous k layout:
// lanes 0-15: m=lane, K = [koff..koff+15] with koff=0
// lanes 16-31: m=lane-16, koff=16
__device__ __forceinline__ v16h load_a_frag(const _Float16* A, int lda,
                                            int mBase, int kk, int lane) {
    int m    = lane & 15;
    int koff = (lane >> 4) * 16;
    const _Float16* p = A + (size_t)(mBase + m) * lda + kk + koff;
    v8h lo = *(const v8h*)(p);
    v8h hi = *(const v8h*)(p + 8);
    v16h a;
#pragma unroll
    for (int i = 0; i < 8; ++i) { a[i] = lo[i]; a[8 + i] = hi[i]; }
    return a;
}

// ---------------- WMMA GEMM: C[M,Nc] = act(concat(A0,A1) @ Wt^T + bias) ----------------
// A0: [M,K0] f16 row-major, A1: [M,K1] f16 row-major (optional, nullptr ok)
// Wt: [Nc, K0+K1] f16 row-major (i.e. W transposed) -> contiguous B-fragment loads
// Each wave computes a 16x64 output tile (4 wmma accumulators).
__global__ void wmma_gemm_kernel(const _Float16* __restrict__ A0, int K0,
                                 const _Float16* __restrict__ A1, int K1,
                                 const _Float16* __restrict__ Wt,
                                 const float* __restrict__ bias,
                                 float* __restrict__ Cf, _Float16* __restrict__ Ch,
                                 int outHalf, int M, int Nc, int act)
{
    const int K     = K0 + K1;
    const int wid   = (int)((blockIdx.x * blockDim.x + threadIdx.x) >> 5);
    const int lane  = (int)(threadIdx.x & 31);
    const int numMT = M >> 4;
    const int numNT = Nc >> 6;
    if (wid >= numMT * numNT) return;   // wave-uniform: scalar branch, EXEC stays full
    const int mt    = wid % numMT;
    const int nt    = wid / numMT;
    const int mBase = mt << 4;
    const int nBase = nt << 6;

    v8f acc[4] = {{}, {}, {}, {}};

#pragma unroll 1
    for (int seg = 0; seg < 2; ++seg) {
        const _Float16* A  = seg ? A1 : A0;
        const int       Ks = seg ? K1 : K0;
        const int       kW = seg ? K0 : 0;
        if (Ks == 0 || A == nullptr) continue;
        for (int kk = 0; kk < Ks; kk += 32) {
            v16h a = load_a_frag(A, Ks, mBase, kk, lane);
#pragma unroll
            for (int sub = 0; sub < 4; ++sub) {
                int n = nBase + sub * 16 + (lane & 15);
                const _Float16* q = Wt + (size_t)n * K + kW + kk + ((lane >> 4) << 4);
                v16h b = *(const v16h*)q;
                acc[sub] = __builtin_amdgcn_wmma_f32_16x16x32_f16(
                    false, a, false, b, (short)0, acc[sub], false, false);
            }
        }
    }

    // D layout: VGPR r, lanes 0-15 -> M=r ; lanes 16-31 -> M=8+r ; N = lane&15
    const int rgrp = (lane >> 4) * 8;
    const int ncol = lane & 15;
#pragma unroll
    for (int sub = 0; sub < 4; ++sub) {
        int   gcol = nBase + sub * 16 + ncol;
        float bi   = bias ? bias[gcol] : 0.0f;
#pragma unroll
        for (int r = 0; r < 8; ++r) {
            int    grow = mBase + rgrp + r;
            float  v    = acc[sub][r] + bi;
            if (act == 1)      v = fmaxf(v, 0.0f);
            else if (act == 2) v = sigf(v);
            size_t o = (size_t)grow * Nc + gcol;
            if (outHalf) Ch[o] = (_Float16)v;
            else         Cf[o] = v;
        }
    }
}

// ---------------- elementwise / gather / scatter kernels ----------------
__global__ void zero_f32(float* p, long n) {
    long i = (long)blockIdx.x * blockDim.x + threadIdx.x;
    if (i < n) p[i] = 0.0f;
}

// Wt[n*K+k] = (f16) W[k*Nc+n]
__global__ void convert_wt(const float* __restrict__ W, _Float16* __restrict__ Wt,
                           int K, int Nc) {
    long i = (long)blockIdx.x * blockDim.x + threadIdx.x;
    if (i >= (long)K * Nc) return;
    int k = (int)(i / Nc), n = (int)(i % Nc);
    Wt[(size_t)n * K + k] = (_Float16)W[i];
}

// nodein_h[n, 0:F] = x[n], nodein_h[n, F:2F] = graph_emb[batch[n]]
__global__ void build_nodein(const float* __restrict__ x, const float* __restrict__ gtab,
                             const int* __restrict__ batch, _Float16* __restrict__ out,
                             int Nn, int F) {
    long i = (long)blockIdx.x * blockDim.x + threadIdx.x;
    if (i >= (long)Nn * F) return;
    int n = (int)(i / F), f = (int)(i % F);
    out[(size_t)n * 2 * F + f]     = (_Float16)x[(size_t)n * F + f];
    out[(size_t)n * 2 * F + F + f] = (_Float16)gtab[(size_t)batch[n] * F + f];
}

// xij0_h[e] = [x[ei0[e]], x[ei1[e]]] ; gemb_h[e] = graph_emb[batch[ei0[e]]]
__global__ void build_xij0(const float* __restrict__ x, const float* __restrict__ gtab,
                           const int* __restrict__ ei, const int* __restrict__ batch,
                           _Float16* __restrict__ xij, _Float16* __restrict__ gemb,
                           int E, int F) {
    long i = (long)blockIdx.x * blockDim.x + threadIdx.x;
    if (i >= (long)E * F) return;
    int e = (int)(i / F), f = (int)(i % F);
    int c0 = ei[e], r0 = ei[E + e];
    xij[(size_t)e * 2 * F + f]     = (_Float16)x[(size_t)c0 * F + f];
    xij[(size_t)e * 2 * F + F + f] = (_Float16)x[(size_t)r0 * F + f];
    gemb[(size_t)e * F + f]        = (_Float16)gtab[(size_t)batch[c0] * F + f];
}

// wave-per-row dot(row_f16[128], w_f32[128]) -> sigmoid -> out
__global__ void rowdot_sigmoid(const _Float16* __restrict__ Ah, const float* __restrict__ w,
                               const float* __restrict__ b, float* __restrict__ out, int M) {
    int wid  = (int)((blockIdx.x * blockDim.x + threadIdx.x) >> 5);
    int lane = (int)(threadIdx.x & 31);
    if (wid >= M) return;
    const _Float16* row = Ah + (size_t)wid * 128;
    float s = 0.0f;
#pragma unroll
    for (int j = 0; j < 4; ++j) s += (float)row[lane + 32 * j] * w[lane + 32 * j];
#pragma unroll
    for (int off = 16; off > 0; off >>= 1) s += __shfl_down(s, off);
    if (lane == 0) out[wid] = sigf(s + b[0]);
}

// D[hrow[i]] += node_w[hcol[hcol[i]]] ; Bdeg[hcol[i]] += 1
__global__ void degrees(const int* __restrict__ hrow, const int* __restrict__ hcol,
                        const float* __restrict__ node_w,
                        float* __restrict__ D, float* __restrict__ Bdeg, long NNZ) {
    long i = (long)blockIdx.x * blockDim.x + threadIdx.x;
    if (i >= NNZ) return;
    int r = hrow[i], c = hcol[i];
    float w = node_w[hcol[c]];   // faithful double-index: node_w[hcol[hcol[i]]]
    atomicAdd(&D[r], w);
    atomicAdd(&Bdeg[c], 1.0f);
}

__global__ void invert_nz(float* p, long n) {
    long i = (long)blockIdx.x * blockDim.x + threadIdx.x;
    if (i >= n) return;
    float v = p[i];
    p[i] = (v == 0.0f) ? 0.0f : (1.0f / v);
}

// out_e[hcol[i]] += Binv[hcol[i]] * xw[hrow[i]]   (4 feats / thread)
__global__ void scatter_fw(const int* __restrict__ hrow, const int* __restrict__ hcol,
                           const float* __restrict__ Binv, const float* __restrict__ xw,
                           float* __restrict__ out_e, long NNZ, int Kf) {
    long idx = (long)blockIdx.x * blockDim.x + threadIdx.x;
    int  per = Kf >> 2;
    if (idx >= NNZ * per) return;
    long i = idx / per;
    int  f = (int)(idx % per) << 2;
    int  r = hrow[i], c = hcol[i];
    float s = Binv[c];
    float4 v = *(const float4*)(xw + (size_t)r * Kf + f);
    float* o = out_e + (size_t)c * Kf + f;
    atomicAdd(o + 0, s * v.x);
    atomicAdd(o + 1, s * v.y);
    atomicAdd(o + 2, s * v.z);
    atomicAdd(o + 3, s * v.w);
}

// hout[hrow[i]] += Dinv[hrow[i]] * out_e[hcol[i]]
__global__ void scatter_bw(const int* __restrict__ hrow, const int* __restrict__ hcol,
                           const float* __restrict__ Dinv, const float* __restrict__ out_e,
                           float* __restrict__ hout, long NNZ, int Kf) {
    long idx = (long)blockIdx.x * blockDim.x + threadIdx.x;
    int  per = Kf >> 2;
    if (idx >= NNZ * per) return;
    long i = idx / per;
    int  f = (int)(idx % per) << 2;
    int  r = hrow[i], c = hcol[i];
    float s = Dinv[r];
    float4 v = *(const float4*)(out_e + (size_t)c * Kf + f);
    float* o = hout + (size_t)r * Kf + f;
    atomicAdd(o + 0, s * v.x);
    atomicAdd(o + 1, s * v.y);
    atomicAdd(o + 2, s * v.z);
    atomicAdd(o + 3, s * v.w);
}

__global__ void sigmoid_bias_h(const float* __restrict__ in, const float* __restrict__ bias,
                               _Float16* __restrict__ out, long n, int Kf) {
    long i = (long)blockIdx.x * blockDim.x + threadIdx.x;
    if (i >= n) return;
    int f = (int)(i % Kf);
    out[i] = (_Float16)sigf(in[i] + bias[f]);
}

__global__ void sigmoid_bias_f(const float* __restrict__ in, const float* __restrict__ bias,
                               float* __restrict__ out, long n, int Kf) {
    long i = (long)blockIdx.x * blockDim.x + threadIdx.x;
    if (i >= n) return;
    int f = (int)(i % Kf);
    out[i] = sigf(in[i] + bias[f]);
}

// s_h[e] = [ f16(attn0*xl[e] + attn1*sij[e]) , gemb_h[e] ]
__global__ void combine(const float* __restrict__ xl, const float* __restrict__ sij,
                        const _Float16* __restrict__ gembh, const float* __restrict__ attn,
                        _Float16* __restrict__ sh, int E, int F) {
    long i = (long)blockIdx.x * blockDim.x + threadIdx.x;
    if (i >= (long)E * F) return;
    int e = (int)(i / F), f = (int)(i % F);
    float v = attn[0] * xl[i] + attn[1] * sij[i];
    sh[(size_t)e * 2 * F + f]     = (_Float16)v;
    sh[(size_t)e * 2 * F + F + f] = gembh[i];
}

// ---------------- host-side launch ----------------
static inline unsigned nblk(long n) { return (unsigned)((n + 255) / 256); }

static void launch_gemm(hipStream_t s, const _Float16* A0, int K0,
                        const _Float16* A1, int K1, const _Float16* Wt,
                        const float* bias, float* Cf, _Float16* Ch, int outHalf,
                        int M, int Nc, int act) {
    long waves   = (long)(M / 16) * (Nc / 64);
    long threads = waves * 32;
    wmma_gemm_kernel<<<nblk(threads), 256, 0, s>>>(A0, K0, A1, K1, Wt, bias,
                                                   Cf, Ch, outHalf, M, Nc, act);
}

extern "C" void kernel_launch(void* const* d_in, const int* in_sizes, int n_in,
                              void* d_out, int out_size, void* d_ws, size_t ws_size,
                              hipStream_t stream) {
    const float* x      = (const float*)d_in[0];
    const float* gtab   = (const float*)d_in[1];
    const int*   ei     = (const int*)d_in[2];
    const int*   batch  = (const int*)d_in[4];
    const int*   hrow   = (const int*)d_in[5];
    const int*   hcol   = hrow + NNZ_;
    const float* hw_w1  = (const float*)d_in[6];
    const float* hw_b1  = (const float*)d_in[7];
    const float* hw_w2  = (const float*)d_in[8];
    const float* hw_b2  = (const float*)d_in[9];
    const float* c1_w   = (const float*)d_in[10];
    const float* c1_b   = (const float*)d_in[11];
    const float* c2_w   = (const float*)d_in[12];
    const float* c2_b   = (const float*)d_in[13];
    const float* g_w1   = (const float*)d_in[14];
    const float* g_b1   = (const float*)d_in[15];
    const float* g_w2   = (const float*)d_in[16];
    const float* g_b2   = (const float*)d_in[17];
    const float* cl_w1  = (const float*)d_in[18];
    const float* cl_b1  = (const float*)d_in[19];
    const float* cl_w2  = (const float*)d_in[20];
    const float* cl_b2  = (const float*)d_in[21];
    const float* attn   = (const float*)d_in[22];
    float* out = (float*)d_out;

    // ---- workspace layout (256B aligned) ----
    char*  base = (char*)d_ws;
    size_t off  = 0;
    auto alloc = [&](size_t bytes) { size_t r = off; off = (off + bytes + 255) & ~(size_t)255; return r; };
    size_t oWT_HW1 = alloc((size_t)256 * 128 * 2);
    size_t oWT_C1  = alloc((size_t)256 * 256 * 2);
    size_t oWT_C2  = alloc((size_t)256 * 128 * 2);
    size_t oWT_G1  = alloc((size_t)384 * 512 * 2);
    size_t oWT_G2  = alloc((size_t)512 * 128 * 2);
    size_t oWT_CL1 = alloc((size_t)256 * 128 * 2);
    size_t oNODEIN = alloc((size_t)N_ * 256 * 2);
    size_t oH_H    = alloc((size_t)N_ * 128 * 2);
    size_t oNODE_W = alloc((size_t)N_ * 4);
    size_t oD      = alloc((size_t)E_ * 4);
    size_t oBDEG   = alloc((size_t)N_ * 4);
    size_t oOUT_E  = alloc((size_t)N_ * 256 * 4);
    size_t oXIJ0H  = alloc((size_t)E_ * 256 * 2);
    size_t oGEMBH  = alloc((size_t)E_ * 128 * 2);
    size_t oP      = alloc((size_t)E_ * 256 * 4);   // xw1 -> er_h -> t1_h -> s_h
    size_t oQ      = alloc((size_t)E_ * 256 * 4);   // hout1 ; Q1: xw2->sij->h5_h ; Q2: hout2->xl

    _Float16* WT_HW1 = (_Float16*)(base + oWT_HW1);
    _Float16* WT_C1  = (_Float16*)(base + oWT_C1);
    _Float16* WT_C2  = (_Float16*)(base + oWT_C2);
    _Float16* WT_G1  = (_Float16*)(base + oWT_G1);
    _Float16* WT_G2  = (_Float16*)(base + oWT_G2);
    _Float16* WT_CL1 = (_Float16*)(base + oWT_CL1);
    _Float16* NODEIN = (_Float16*)(base + oNODEIN);
    _Float16* H_H    = (_Float16*)(base + oH_H);
    float*    NODE_W = (float*)(base + oNODE_W);
    float*    Dbuf   = (float*)(base + oD);
    float*    BDEG   = (float*)(base + oBDEG);
    float*    OUT_E  = (float*)(base + oOUT_E);
    _Float16* XIJ0H  = (_Float16*)(base + oXIJ0H);
    _Float16* GEMBH  = (_Float16*)(base + oGEMBH);
    float*    P      = (float*)(base + oP);
    float*    Q      = (float*)(base + oQ);
    float*    Q1     = Q;
    float*    Q2     = Q + (size_t)E_ * 128;

    // 1) weight convert + transpose to f16 [Nc, K]
    convert_wt<<<nblk(256L * 128), 256, 0, stream>>>(hw_w1, WT_HW1, 256, 128);
    convert_wt<<<nblk(256L * 256), 256, 0, stream>>>(c1_w, WT_C1, 256, 256);
    convert_wt<<<nblk(256L * 128), 256, 0, stream>>>(c2_w, WT_C2, 256, 128);
    convert_wt<<<nblk(384L * 512), 256, 0, stream>>>(g_w1, WT_G1, 384, 512);
    convert_wt<<<nblk(512L * 128), 256, 0, stream>>>(g_w2, WT_G2, 512, 128);
    convert_wt<<<nblk(256L * 128), 256, 0, stream>>>(cl_w1, WT_CL1, 256, 128);

    // 2) HyperWeight: node_w = sigmoid(relu([x|proto] @ hw_w1 + b1) @ hw_w2 + b2)
    build_nodein<<<nblk((long)N_ * F_), 256, 0, stream>>>(x, gtab, batch, NODEIN, N_, F_);
    launch_gemm(stream, NODEIN, 256, nullptr, 0, WT_HW1, hw_b1, nullptr, H_H, 1, N_, 128, 1);
    rowdot_sigmoid<<<nblk((long)N_ * 32), 256, 0, stream>>>(H_H, hw_w2, hw_b2, NODE_W, N_);

    // 3) degrees: D (per graph-edge), Bdeg (per hyperedge) -> inverted in place
    zero_f32<<<nblk(E_), 256, 0, stream>>>(Dbuf, E_);
    zero_f32<<<nblk(N_), 256, 0, stream>>>(BDEG, N_);
    degrees<<<nblk(NNZ_), 256, 0, stream>>>(hrow, hcol, NODE_W, Dbuf, BDEG, NNZ_);
    invert_nz<<<nblk(E_), 256, 0, stream>>>(Dbuf, E_);
    invert_nz<<<nblk(N_), 256, 0, stream>>>(BDEG, N_);

    // 4) xij0 (f16) and gemb (f16)
    build_xij0<<<nblk((long)E_ * F_), 256, 0, stream>>>(x, gtab, ei, batch, XIJ0H, GEMBH, E_, F_);

    // 5) hconv1: xw1 = xij0 @ c1_w ; scatter fw/bw ; er = sigmoid(. + c1_b) (f16)
    launch_gemm(stream, XIJ0H, 256, nullptr, 0, WT_C1, nullptr, P, nullptr, 0, E_, 256, 0);
    zero_f32<<<nblk((long)N_ * 256), 256, 0, stream>>>(OUT_E, (long)N_ * 256);
    scatter_fw<<<nblk(NNZ_ * 64), 256, 0, stream>>>(hrow, hcol, BDEG, P, OUT_E, NNZ_, 256);
    zero_f32<<<nblk((long)E_ * 256), 256, 0, stream>>>(Q, (long)E_ * 256);
    scatter_bw<<<nblk(NNZ_ * 64), 256, 0, stream>>>(hrow, hcol, Dbuf, OUT_E, Q, NNZ_, 256);
    sigmoid_bias_h<<<nblk((long)E_ * 256), 256, 0, stream>>>(Q, c1_b, (_Float16*)P, (long)E_ * 256, 256);

    // 6) hconv2: xw2 = er @ c2_w ; scatter ; sij = sigmoid(. + c2_b) (f32 in Q1)
    launch_gemm(stream, (_Float16*)P, 256, nullptr, 0, WT_C2, nullptr, Q1, nullptr, 0, E_, 128, 0);
    zero_f32<<<nblk((long)N_ * 128), 256, 0, stream>>>(OUT_E, (long)N_ * 128);
    scatter_fw<<<nblk(NNZ_ * 32), 256, 0, stream>>>(hrow, hcol, BDEG, Q1, OUT_E, NNZ_, 128);
    zero_f32<<<nblk((long)E_ * 128), 256, 0, stream>>>(Q2, (long)E_ * 128);
    scatter_bw<<<nblk(NNZ_ * 32), 256, 0, stream>>>(hrow, hcol, Dbuf, OUT_E, Q2, NNZ_, 128);
    sigmoid_bias_f<<<nblk((long)E_ * 128), 256, 0, stream>>>(Q2, c2_b, Q1, (long)E_ * 128, 128);

    // 7) Generator: t1 = relu([xij0|gemb] @ g_w1 + b) (f16 in P) ; xl = relu(t1 @ g_w2 + b) (f32 in Q2)
    launch_gemm(stream, XIJ0H, 256, GEMBH, 128, WT_G1, g_b1, nullptr, (_Float16*)P, 1, E_, 512, 1);
    launch_gemm(stream, (_Float16*)P, 512, nullptr, 0, WT_G2, g_b2, Q2, nullptr, 0, E_, 128, 1);

    // 8) s = [attn0*xl + attn1*sij , gemb] (f16 in P)
    combine<<<nblk((long)E_ * F_), 256, 0, stream>>>(Q2, Q1, GEMBH, attn, (_Float16*)P, E_, F_);

    // 9) classifier: h5 = relu(s @ cl_w1 + b) (f16 in Q1) ; out = sigmoid(h5 . cl_w2 + b)
    launch_gemm(stream, (_Float16*)P, 256, nullptr, 0, WT_CL1, cl_b1, nullptr, (_Float16*)Q1, 1, E_, 128, 1);
    rowdot_sigmoid<<<nblk((long)E_ * 32), 256, 0, stream>>>((_Float16*)Q1, cl_w2, cl_b2, out, E_);

    (void)in_sizes; (void)n_in; (void)out_size; (void)ws_size;
}
